// NodeLevelPromptRefiner_57019985821823
// MI455X (gfx1250) — matrix operand order repaired and checked by reading, not compiled, not run on hardware
//
#include <hip/hip_runtime.h>
#include <hip/hip_bf16.h>

typedef __attribute__((ext_vector_type(16))) __bf16 v16bf;
typedef __attribute__((ext_vector_type(8)))  __bf16 v8bf;
typedef __attribute__((ext_vector_type(8)))  float  v8f;
typedef __attribute__((ext_vector_type(4)))  int    v4i;

#define NN       100000   // nodes
#define DIM      512      // node/out dim
#define KK1      1024     // concat fan-in
#define BM       96       // rows per block
#define NTH      384      // 12 waves: 3 M-waves x 4 N-waves
#define KC       32       // K chunk per stage (one WMMA k-step)
#define KCP      40       // padded K stride (bf16), 80B = 16B-aligned
#define HP       520      // padded Hs row stride (bf16), 1040B = 16B-aligned

#define AS1 __attribute__((address_space(1)))
#define AS3 __attribute__((address_space(3)))

#if __has_builtin(__builtin_amdgcn_global_load_async_to_lds_b128) && \
    __has_builtin(__builtin_amdgcn_s_wait_asynccnt)
#define HAS_ASYNC 1
#else
#define HAS_ASYNC 0
#endif

// ---- prep: W1/W2 f32 row-major -> bf16 transposed (n-major) in workspace ----
__global__ void __launch_bounds__(256)
prep_weights(const float* __restrict__ W1, const float* __restrict__ W2,
             __bf16* __restrict__ wt)
{
    int idx = blockIdx.x * 256 + threadIdx.x;
    const int total1 = DIM * KK1;          // Wt1: [DIM][KK1]
    const int total2 = DIM * DIM;          // Wt2: [DIM][DIM]
    if (idx < total1) {
        int n = idx / KK1, k = idx - n * KK1;
        wt[idx] = (__bf16)W1[k * DIM + n];
    } else if (idx < total1 + total2) {
        int i2 = idx - total1;
        int n = i2 / DIM, k = i2 - n * DIM;
        wt[idx] = (__bf16)W2[k * DIM + n];
    }
}

// stage pre-transposed bf16 weights: Wt[n][k0..k0+31] -> Bs[n][0..31]
__device__ __forceinline__ void stage_wt(const __bf16* __restrict__ wt, int kt,
                                         int k0, __bf16* Bs, int tid)
{
#if HAS_ASYNC
    // 512 rows * 4 x 16B units, async DMA into LDS (ASYNCcnt path)
    for (int u = tid; u < DIM * 4; u += NTH) {
        int n = u >> 2;
        int g = u & 3;
        __builtin_amdgcn_global_load_async_to_lds_b128(
            (AS1 v4i*)(wt + n * kt + k0 + g * 8),
            (AS3 v4i*)(void*)(Bs + n * KCP + g * 8),
            0, 0);
    }
    __builtin_amdgcn_s_wait_asynccnt(0);
#else
    for (int u = tid; u < DIM * 4; u += NTH) {
        int n = u >> 2;
        int g = u & 3;
        *(v8bf*)(Bs + n * KCP + g * 8) = *(const v8bf*)(wt + n * kt + k0 + g * 8);
    }
#endif
}

// fallback: stage f32 weights with transpose + convert
__device__ __forceinline__ void stage_w_f32(const float* __restrict__ W, int k0,
                                            __bf16* Bs, int tid)
{
    for (int i = tid; i < KC * DIM; i += NTH) {
        int k = i >> 9;
        int n = i & 511;
        Bs[n * KCP + k] = (__bf16)W[(k0 + k) * DIM + n];
    }
}

__global__ void __launch_bounds__(NTH)
fused_prompt_mlp(const float* __restrict__ node,
                 const float* __restrict__ prompt,
                 const int*   __restrict__ bidx,
                 const float* __restrict__ W1,
                 const float* __restrict__ bias1,
                 const float* __restrict__ W2,
                 const float* __restrict__ bias2,
                 const __bf16* __restrict__ Wt1,   // nullptr if no workspace
                 const __bf16* __restrict__ Wt2,
                 float* __restrict__ out)
{
    __shared__ __bf16 As[BM * KCP];   //  7,680 B
    __shared__ __bf16 Bs[DIM * KCP];  // 40,960 B
    __shared__ __bf16 Hs[BM * HP];    // 99,840 B
    __shared__ int    sB[BM];

    const int tid  = threadIdx.x;
    const int lane = tid & 31;
    const int wave = tid >> 5;        // 0..11
    const int wm   = wave % 3;        // 3 M-waves of 32 rows
    const int wn   = wave / 3;        // 4 N-waves of 128 cols
    const int rm   = wm * 32;
    const int cn   = wn * 128;
    const int hf   = lane >> 4;
    const int l15  = lane & 15;
    const int row0 = blockIdx.x * BM;
    const bool useWt = (Wt1 != nullptr);

    if (tid < BM) {
        int gr = row0 + tid; if (gr >= NN) gr = NN - 1;
        sB[tid] = bidx[gr];
    }

    v8f acc[2][8];
    #pragma unroll
    for (int mi = 0; mi < 2; ++mi)
        #pragma unroll
        for (int nt = 0; nt < 8; ++nt)
            acc[mi][nt] = (v8f){0.f,0.f,0.f,0.f,0.f,0.f,0.f,0.f};

    // ============================ Layer 1 ============================
    for (int k0 = 0; k0 < KK1; k0 += KC) {
        __syncthreads();                       // prior compute done with Bs/As
        if (useWt) {
            if (k0 + KC < KK1) __builtin_prefetch(&Wt1[k0 + KC], 0, 0);
            stage_wt(Wt1, KK1, k0, Bs, tid);
        } else {
            if (k0 + KC < KK1) __builtin_prefetch(&W1[(k0 + KC) * DIM], 0, 0);
            stage_w_f32(W1, k0, Bs, tid);
        }
        // stage combined A chunk (always converts f32 -> bf16)
        const bool isPrompt = (k0 >= DIM);     // uniform per chunk
        for (int i = tid; i < BM * KC; i += NTH) {
            int r = i >> 5;
            int k = i & 31;
            int gr = row0 + r; if (gr >= NN) gr = NN - 1;
            float v = isPrompt ? prompt[sB[r] * DIM + (k0 - DIM) + k]
                               : node[gr * DIM + k0 + k];
            As[r * KCP + k] = (__bf16)v;
        }
        __syncthreads();

        v16bf afrag[2];
        #pragma unroll
        for (int mi = 0; mi < 2; ++mi) {
            const __bf16* ap = &As[(rm + mi * 16 + l15) * KCP + 8 * hf];
            v8bf lo = *(const v8bf*)ap;
            v8bf hi = *(const v8bf*)(ap + 16);
            #pragma unroll
            for (int e = 0; e < 8; ++e) { afrag[mi][e] = lo[e]; afrag[mi][e + 8] = hi[e]; }
        }
        #pragma unroll
        for (int nt = 0; nt < 8; ++nt) {
            const __bf16* bp = &Bs[(cn + nt * 16 + l15) * KCP + 8 * hf];
            v8bf lo = *(const v8bf*)bp;
            v8bf hi = *(const v8bf*)(bp + 16);
            v16bf bfrag;
            #pragma unroll
            for (int e = 0; e < 8; ++e) { bfrag[e] = lo[e]; bfrag[e + 8] = hi[e]; }
            acc[0][nt] = __builtin_amdgcn_wmma_f32_16x16x32_bf16(
                false, afrag[0], false, bfrag, (short)0, acc[0][nt], false, false);
            acc[1][nt] = __builtin_amdgcn_wmma_f32_16x16x32_bf16(
                false, afrag[1], false, bfrag, (short)0, acc[1][nt], false, false);
        }
    }

    // epilogue 1: bias + ReLU -> Hs (bf16); reset accumulators
    #pragma unroll
    for (int nt = 0; nt < 8; ++nt) {
        int col = cn + nt * 16 + l15;
        float b = bias1[col];
        #pragma unroll
        for (int mi = 0; mi < 2; ++mi) {
            #pragma unroll
            for (int e = 0; e < 8; ++e) {
                int row = rm + mi * 16 + e + 8 * hf;
                float x = acc[mi][nt][e] + b;
                x = x > 0.f ? x : 0.f;
                Hs[row * HP + col] = (__bf16)x;
            }
            acc[mi][nt] = (v8f){0.f,0.f,0.f,0.f,0.f,0.f,0.f,0.f};
        }
    }
    __syncthreads();

    // ============================ Layer 2 ============================
    for (int k0 = 0; k0 < DIM; k0 += KC) {
        __syncthreads();
        if (useWt) {
            if (k0 + KC < DIM) __builtin_prefetch(&Wt2[k0 + KC], 0, 0);
            stage_wt(Wt2, DIM, k0, Bs, tid);
        } else {
            if (k0 + KC < DIM) __builtin_prefetch(&W2[(k0 + KC) * DIM], 0, 0);
            stage_w_f32(W2, k0, Bs, tid);
        }
        __syncthreads();

        v16bf afrag[2];
        #pragma unroll
        for (int mi = 0; mi < 2; ++mi) {
            const __bf16* ap = &Hs[(rm + mi * 16 + l15) * HP + k0 + 8 * hf];
            v8bf lo = *(const v8bf*)ap;
            v8bf hi = *(const v8bf*)(ap + 16);
            #pragma unroll
            for (int e = 0; e < 8; ++e) { afrag[mi][e] = lo[e]; afrag[mi][e + 8] = hi[e]; }
        }
        #pragma unroll
        for (int nt = 0; nt < 8; ++nt) {
            const __bf16* bp = &Bs[(cn + nt * 16 + l15) * KCP + 8 * hf];
            v8bf lo = *(const v8bf*)bp;
            v8bf hi = *(const v8bf*)(bp + 16);
            v16bf bfrag;
            #pragma unroll
            for (int e = 0; e < 8; ++e) { bfrag[e] = lo[e]; bfrag[e + 8] = hi[e]; }
            acc[0][nt] = __builtin_amdgcn_wmma_f32_16x16x32_bf16(
                false, afrag[0], false, bfrag, (short)0, acc[0][nt], false, false);
            acc[1][nt] = __builtin_amdgcn_wmma_f32_16x16x32_bf16(
                false, afrag[1], false, bfrag, (short)0, acc[1][nt], false, false);
        }
    }

    // epilogue 2: bias2 + store f32 to global
    #pragma unroll
    for (int nt = 0; nt < 8; ++nt) {
        int col = cn + nt * 16 + l15;
        float b = bias2[col];
        #pragma unroll
        for (int mi = 0; mi < 2; ++mi)
            #pragma unroll
            for (int e = 0; e < 8; ++e) {
                int row = rm + mi * 16 + e + 8 * hf;
                int gr  = row0 + row;
                if (gr < NN)
                    out[gr * DIM + col] = acc[mi][nt][e] + b;
            }
    }
}

extern "C" void kernel_launch(void* const* d_in, const int* in_sizes, int n_in,
                              void* d_out, int out_size, void* d_ws, size_t ws_size,
                              hipStream_t stream) {
    (void)in_sizes; (void)n_in; (void)out_size;
    const float* node   = (const float*)d_in[0];
    const float* prompt = (const float*)d_in[1];
    const int*   bidx   = (const int*)  d_in[2];
    const float* W1     = (const float*)d_in[3];
    const float* bias1  = (const float*)d_in[4];
    const float* W2     = (const float*)d_in[5];
    const float* bias2  = (const float*)d_in[6];
    float* out = (float*)d_out;

    const size_t wtBytes = (size_t)(DIM * KK1 + DIM * DIM) * sizeof(__bf16); // 1.5 MB
    const __bf16* Wt1 = nullptr;
    const __bf16* Wt2 = nullptr;
    if (d_ws && ws_size >= wtBytes) {
        __bf16* wt = (__bf16*)d_ws;
        const int totalW = DIM * KK1 + DIM * DIM;
        prep_weights<<<(totalW + 255) / 256, 256, 0, stream>>>(W1, W2, wt);
        Wt1 = wt;
        Wt2 = wt + DIM * KK1;
    }

    const int grid = (NN + BM - 1) / BM;   // 1042 blocks
    fused_prompt_mlp<<<grid, NTH, 0, stream>>>(node, prompt, bidx, W1, bias1,
                                               W2, bias2, Wt1, Wt2, out);
}
